// SpikeNet_26465588478212
// MI455X (gfx1250) — compile-verified, matrix operand order checked by reading, not run on hardware
//
#include <hip/hip_runtime.h>
#include <hip/hip_bf16.h>
#include <cstdint>

// ---------------------------------------------------------------------------
// SpikeNet on MI455X (gfx1250).
//
// TAU==1.0 makes the LIF state vanish (v <- z every step), so each timestep is
// an independent  gather -> GEMM -> threshold  pipeline. We:
//   * convert x (500000x128 f32, 256MB) to f16 once (128MB -> fully L2
//     resident on the 192MB L2; all random gathers then hit L2, not HBM)
//   * run layer0 as a 98304x256x128 f16 WMMA GEMM (A = [self | neighbor-mean]
//     built in LDS, B = concat(Wl0;Wr0)^T hoisted into VGPRs per wave)
//   * run layer1 as 16384x256x64, projection as 16384x320x40 (padded to 48)
// All matrix math uses v_wmma_f32_16x16x32_f16 (f32 accumulate).
// ---------------------------------------------------------------------------

typedef _Float16 v16h __attribute__((ext_vector_type(16)));
typedef float    v8f  __attribute__((ext_vector_type(8)));
typedef _Float16 h4   __attribute__((ext_vector_type(4)));

namespace cfg {
constexpr int NUM_NODES = 500000;
constexpr int F         = 128;
constexpr int N         = 16384;   // root nodes
constexpr int S0        = 5;
constexpr int S1        = 2;
constexpr int T         = 5;
constexpr int H0        = 128;
constexpr int H1        = 64;
constexpr int C         = 40;
constexpr int M0        = N + N * S0;        // 98304 rows in layer 0
constexpr int K01       = 2 * F;             // 256 (self | neigh concat)
constexpr int KP        = T * H1;            // 320
constexpr int CP        = 48;                // C padded to 3 n-tiles
}  // namespace cfg

// ---------------------------------------------------------------------------
// Fragment loader for 16-bit A/B operands of V_WMMA_*_16X16X32 (wave32).
// Row-major source with row stride `stride` (halves). ISA layout (CDNA5 ISA):
//   lanes 0-15 : row = lane,     VGPR v holds K = {2v, 2v+1} (v<4)  /
//                                         K = {16+2(v-4), ...} (v>=4)
//   lanes 16-31: row = lane-16,  same pattern shifted by K+8
// K pairs are contiguous in memory -> dword-mergeable loads.
// ---------------------------------------------------------------------------
__device__ __forceinline__ v16h frag_load(const _Float16* __restrict__ base,
                                          int row, int stride, int koff,
                                          int lane) {
  const _Float16* p = base + (size_t)row * stride + koff + ((lane & 16) ? 8 : 0);
  v16h f;
#pragma unroll
  for (int v = 0; v < 8; ++v) {
    const int k = (v < 4) ? (2 * v) : (8 + 2 * v);  // v>=4: 16 + 2*(v-4)
    f[2 * v]     = p[k];
    f[2 * v + 1] = p[k + 1];
  }
  return f;
}

__device__ __forceinline__ v8f wmma16(v16h a, v16h b, v8f c) {
  return __builtin_amdgcn_wmma_f32_16x16x32_f16(false, a, false, b,
                                                (short)0, c, false, false);
}

// ---------------------------------------------------------------------------
// x (f32) -> xh (f16), vectorized 4-wide.  64M elements.
// ---------------------------------------------------------------------------
__global__ void cvt_x_kernel(const float* __restrict__ x,
                             _Float16* __restrict__ xh, int n4) {
  int i = blockIdx.x * blockDim.x + threadIdx.x;
  if (i >= n4) return;
  const float4 v = reinterpret_cast<const float4*>(x)[i];
  h4 h = {(_Float16)v.x, (_Float16)v.y, (_Float16)v.z, (_Float16)v.w};
  reinterpret_cast<h4*>(xh)[i] = h;
}

// ---------------------------------------------------------------------------
// Build transposed f16 weight blocks:
//   w0t[n][k] (128x256) = k<128 ? Wl0[k][n] : Wr0[k-128][n]
//   w1t[n][k] ( 64x256) = k<128 ? Wl1[k][n] : Wr1[k-128][n]
//   wpt[n][k] ( 48x320) = n<40  ? Wp[k][n]  : 0
// ---------------------------------------------------------------------------
__global__ void prep_w_kernel(const float* __restrict__ Wl0,
                              const float* __restrict__ Wr0,
                              const float* __restrict__ Wl1,
                              const float* __restrict__ Wr1,
                              const float* __restrict__ Wp,
                              _Float16* __restrict__ w0t,
                              _Float16* __restrict__ w1t,
                              _Float16* __restrict__ wpt) {
  using namespace cfg;
  const int tid = blockIdx.x * blockDim.x + threadIdx.x;
  const int stride = gridDim.x * blockDim.x;
  for (int idx = tid; idx < H0 * K01; idx += stride) {
    const int n = idx >> 8, k = idx & 255;
    const float v = (k < F) ? Wl0[k * H0 + n] : Wr0[(k - F) * H0 + n];
    w0t[idx] = (_Float16)v;
  }
  for (int idx = tid; idx < H1 * K01; idx += stride) {
    const int n = idx >> 8, k = idx & 255;
    const float v = (k < H0) ? Wl1[k * H1 + n] : Wr1[(k - H0) * H1 + n];
    w1t[idx] = (_Float16)v;
  }
  for (int idx = tid; idx < CP * KP; idx += stride) {
    const int n = idx / KP, k = idx % KP;
    wpt[idx] = (n < C) ? (_Float16)Wp[k * C + n] : (_Float16)0.0f;
  }
}

// ---------------------------------------------------------------------------
// Layer 0: spk0 = ( [self0|neigh0] @ [Wl0;Wr0] + b >= 1 )   (98304 x 128)
// Block: 128 thr (4 waves), 8 M-tiles per block. B fragments held in VGPRs.
// ---------------------------------------------------------------------------
__global__ __launch_bounds__(128) void l0_kernel(
    const _Float16* __restrict__ xh, const int* __restrict__ nodes,
    const int* __restrict__ nbr1, const int* __restrict__ nbr2,
    const float* __restrict__ bl0, const float* __restrict__ br0,
    const _Float16* __restrict__ w0t, _Float16* __restrict__ spk0, int t) {
  using namespace cfg;
  __shared__ _Float16 As[16 * K01];  // 8 KB
  const int tid  = threadIdx.x;
  const int wave = tid >> 5;
  const int lane = tid & 31;
  const int nl   = lane & 15;
  const int mo   = (lane & 16) ? 8 : 0;
  const int n0   = (wave * 2 + 0) * 16 + nl;
  const int n1   = (wave * 2 + 1) * 16 + nl;

  // Hoist this wave's two B column-tiles (K=256) into registers: reused for
  // all 8 M-tiles of this block.
  v16h bf[2][8];
#pragma unroll
  for (int j = 0; j < 2; ++j)
#pragma unroll
    for (int kb = 0; kb < 8; ++kb)
      bf[j][kb] = frag_load(w0t, (wave * 2 + j) * 16 + nl, K01, kb * 32, lane);

  const float bias0 = bl0[n0] + br0[n0];
  const float bias1 = bl0[n1] + br0[n1];

  for (int mt = 0; mt < 8; ++mt) {
    const int mbase = (blockIdx.x * 8 + mt) * 16;
    __syncthreads();  // protect As against readers of previous tile
    {  // build A tile: 16 rows x [ self(128) | neighbor-mean(128) ]
      const int r  = tid >> 3;        // 0..15
      const int cb = (tid & 7) * 16;  // 16-col slice
      const int R  = mbase + r;
      const _Float16* srow;
      const int* nidx;
      int nn;
      float scale;
      if (R < N) {  // root node rows
        srow  = xh + (size_t)nodes[R] * F;
        nidx  = nbr1 + (size_t)t * N * S0 + (size_t)R * S0;
        nn    = S0;
        scale = 1.0f / S0;
      } else {  // 1-hop rows
        const int Rp = R - N;
        srow  = xh + (size_t)nbr1[(size_t)t * N * S0 + Rp] * F;
        nidx  = nbr2 + (size_t)t * N * S0 * S1 + (size_t)Rp * S1;
        nn    = S1;
        scale = 1.0f / S1;
      }
      float acc[16];
#pragma unroll
      for (int c = 0; c < 16; ++c) {
        As[r * K01 + cb + c] = srow[cb + c];
        acc[c] = 0.0f;
      }
      for (int i = 0; i < nn; ++i) {
        const _Float16* nrow = xh + (size_t)nidx[i] * F;
#pragma unroll
        for (int c = 0; c < 16; ++c) acc[c] += (float)nrow[cb + c];
      }
#pragma unroll
      for (int c = 0; c < 16; ++c)
        As[r * K01 + F + cb + c] = (_Float16)(acc[c] * scale);
    }
    __syncthreads();

    v8f c0 = {}, c1 = {};
#pragma unroll
    for (int kb = 0; kb < 8; ++kb) {
      const v16h a = frag_load(As, nl, K01, kb * 32, lane);
      c0 = wmma16(a, bf[0][kb], c0);
      c1 = wmma16(a, bf[1][kb], c1);
    }
    // bias + spike threshold + f16 store  (D layout: VGPR g -> M = g + mo)
#pragma unroll
    for (int g = 0; g < 8; ++g) {
      const size_t row = (size_t)(mbase + mo + g);
      const float z0 = c0[g] + bias0;
      const float z1 = c1[g] + bias1;
      spk0[row * H0 + n0] = (_Float16)(z0 >= 1.0f ? 1.0f : 0.0f);
      spk0[row * H0 + n1] = (_Float16)(z1 >= 1.0f ? 1.0f : 0.0f);
    }
  }
}

// ---------------------------------------------------------------------------
// Layer 1: spk1 = ( [spk0[:N] | mean_s0(spk0[N:])] @ [Wl1;Wr1] + b >= 1 )
// Output written into spikes[:, t*64 : (t+1)*64].   (16384 x 64)
// ---------------------------------------------------------------------------
__global__ __launch_bounds__(128) void l1_kernel(
    const _Float16* __restrict__ spk0, const float* __restrict__ bl1,
    const float* __restrict__ br1, const _Float16* __restrict__ w1t,
    _Float16* __restrict__ spikes, int t) {
  using namespace cfg;
  __shared__ _Float16 As[16 * K01];
  const int tid  = threadIdx.x;
  const int wave = tid >> 5;  // 4 waves x 16 cols = 64
  const int lane = tid & 31;
  const int nl   = lane & 15;
  const int mo   = (lane & 16) ? 8 : 0;
  const int n    = wave * 16 + nl;

  v16h bf[8];
#pragma unroll
  for (int kb = 0; kb < 8; ++kb)
    bf[kb] = frag_load(w1t, wave * 16 + nl, K01, kb * 32, lane);
  const float bias = bl1[n] + br1[n];

  for (int mt = 0; mt < 8; ++mt) {
    const int mbase = (blockIdx.x * 8 + mt) * 16;
    __syncthreads();
    {
      const int r  = tid >> 3;
      const int cb = (tid & 7) * 16;
      const int R  = mbase + r;
      const _Float16* srow = spk0 + (size_t)R * H0;
      float acc[16];
#pragma unroll
      for (int c = 0; c < 16; ++c) {
        As[r * K01 + cb + c] = srow[cb + c];
        acc[c] = 0.0f;
      }
#pragma unroll
      for (int i = 0; i < S0; ++i) {
        const _Float16* nrow = spk0 + (size_t)(N + R * S0 + i) * H0;
#pragma unroll
        for (int c = 0; c < 16; ++c) acc[c] += (float)nrow[cb + c];
      }
#pragma unroll
      for (int c = 0; c < 16; ++c)
        As[r * K01 + H0 + cb + c] = (_Float16)(acc[c] * (1.0f / S0));
    }
    __syncthreads();

    v8f c = {};
#pragma unroll
    for (int kb = 0; kb < 8; ++kb) {
      const v16h a = frag_load(As, nl, K01, kb * 32, lane);
      c = wmma16(a, bf[kb], c);
    }
#pragma unroll
    for (int g = 0; g < 8; ++g) {
      const float z = c[g] + bias;
      spikes[(size_t)(mbase + mo + g) * KP + t * H1 + n] =
          (_Float16)(z >= 1.0f ? 1.0f : 0.0f);
    }
  }
}

// ---------------------------------------------------------------------------
// Projection: out = spikes(16384x320) @ Wp(320x40) + bp   (f32 out)
// 3 waves cover 48 padded columns; lanes with n>=40 skip the store.
// ---------------------------------------------------------------------------
__global__ __launch_bounds__(96) void proj_kernel(
    const _Float16* __restrict__ spikes, const _Float16* __restrict__ wpt,
    const float* __restrict__ bp, float* __restrict__ out) {
  using namespace cfg;
  const int tid  = threadIdx.x;
  const int wave = tid >> 5;  // 0..2
  const int lane = tid & 31;
  const int nl   = lane & 15;
  const int mo   = (lane & 16) ? 8 : 0;
  const int n    = wave * 16 + nl;

  v16h bf[10];
#pragma unroll
  for (int kb = 0; kb < 10; ++kb)
    bf[kb] = frag_load(wpt, wave * 16 + nl, KP, kb * 32, lane);

  for (int mt = 0; mt < 4; ++mt) {
    const int mbase = (blockIdx.x * 4 + mt) * 16;
    const _Float16* abase = spikes + (size_t)mbase * KP;
    v8f c = {};
#pragma unroll
    for (int kb = 0; kb < 10; ++kb) {
      const v16h a = frag_load(abase, nl, KP, kb * 32, lane);
      c = wmma16(a, bf[kb], c);
    }
    if (n < C) {
      const float bias = bp[n];
#pragma unroll
      for (int g = 0; g < 8; ++g)
        out[(size_t)(mbase + mo + g) * C + n] = c[g] + bias;
    }
  }
}

// ---------------------------------------------------------------------------
// Host launch.  Workspace layout (~164 MB):
//   xh     : 500000*128 f16       128.0 MB
//   spk0   :  98304*128 f16        25.2 MB
//   spikes :  16384*320 f16        10.5 MB
//   w0t/w1t/wpt                   ~0.11 MB
// ---------------------------------------------------------------------------
extern "C" void kernel_launch(void* const* d_in, const int* in_sizes, int n_in,
                              void* d_out, int out_size, void* d_ws,
                              size_t ws_size, hipStream_t stream) {
  using namespace cfg;
  const float* x     = (const float*)d_in[0];
  const int*   nodes = (const int*)d_in[1];
  const int*   nbr1  = (const int*)d_in[2];
  const int*   nbr2  = (const int*)d_in[3];
  const float* Wl0   = (const float*)d_in[4];
  const float* bl0   = (const float*)d_in[5];
  const float* Wr0   = (const float*)d_in[6];
  const float* br0   = (const float*)d_in[7];
  const float* Wl1   = (const float*)d_in[8];
  const float* bl1   = (const float*)d_in[9];
  const float* Wr1   = (const float*)d_in[10];
  const float* br1   = (const float*)d_in[11];
  const float* Wp    = (const float*)d_in[12];
  const float* bp    = (const float*)d_in[13];
  float* out = (float*)d_out;

  char* ws = (char*)d_ws;
  size_t off = 0;
  auto carve = [&](size_t bytes) {
    void* p = ws + off;
    off = (off + bytes + 255) & ~(size_t)255;
    return p;
  };
  _Float16* xh     = (_Float16*)carve((size_t)NUM_NODES * F * 2);
  _Float16* spk0   = (_Float16*)carve((size_t)M0 * H0 * 2);
  _Float16* spikes = (_Float16*)carve((size_t)N * KP * 2);
  _Float16* w0t    = (_Float16*)carve((size_t)H0 * K01 * 2);
  _Float16* w1t    = (_Float16*)carve((size_t)H1 * K01 * 2);
  _Float16* wpt    = (_Float16*)carve((size_t)CP * KP * 2);

  const int n4 = NUM_NODES * F / 4;  // 16,000,000
  cvt_x_kernel<<<(n4 + 255) / 256, 256, 0, stream>>>(x, xh, n4);
  prep_w_kernel<<<64, 256, 0, stream>>>(Wl0, Wr0, Wl1, Wr1, Wp, w0t, w1t, wpt);

  for (int t = 0; t < T; ++t) {
    l0_kernel<<<M0 / (16 * 8), 128, 0, stream>>>(xh, nodes, nbr1, nbr2, bl0,
                                                 br0, w0t, spk0, t);
    l1_kernel<<<N / (16 * 8), 128, 0, stream>>>(spk0, bl1, br1, w1t, spikes, t);
  }
  proj_kernel<<<N / (16 * 4), 96, 0, stream>>>(spikes, wpt, bp, out);
}
